// MPNEncoder_47510928228862
// MI455X (gfx1250) — matrix operand order repaired and checked by reading, not compiled, not run on hardware
//
#include <hip/hip_runtime.h>
#include <hip/hip_bf16.h>

// ---------------- types ----------------
typedef __bf16 v16bf __attribute__((ext_vector_type(16)));
typedef __bf16 v8bf  __attribute__((ext_vector_type(8)));
typedef __bf16 v4bf  __attribute__((ext_vector_type(4)));
typedef float  v8f   __attribute__((ext_vector_type(8)));
typedef float  v4f   __attribute__((ext_vector_type(4)));

// ---------------- problem constants (match reference) ----------------
#define N_ATOMS   200000
#define N_BONDS   400000
#define ATOM_FDIM 133
#define BOND_FDIM 147
#define HIDDEN    300
#define N_MOLS    10000
#define MAX_NB    6

// padded GEMM dims (K padded to multiple of 32 for 16x16x32 WMMA)
#define KP_I 160   // 147 -> 160
#define KP_H 320   // 300 -> 320
#define KP_O 448   // 433 -> 448
#define NPAD 320   // 300 -> 320 output columns = 20 tiles -> 5 uniform groups of 4
                   // (tiles 19 and part of 18 are pad; discarded at store)

// =====================================================================
// WMMA GEMM:  out = epilogue( A[M x KPAD](bf16) @ Wt^T ),  Wt is [NPAD x KPAD]
// stored N-major (row n = output column n, contiguous in K) -> B fragments
// are contiguous 16B loads.  One wave -> 2 M-tiles x 4 N-tiles (8 WMMA/kstep),
// fully unrolled, branch-free inner loop (all tile indices compile-time valid).
// MODE 0: outRaw = acc; outRelu = relu(acc)              (W_i GEMM)
// MODE 1: outRelu = relu(inpAdd + acc)                   (W_h GEMM)
// MODE 2: outRelu = relu(acc + bias[col])                (W_o GEMM)
// =====================================================================
template <int KPAD, int MODE>
__global__ __launch_bounds__(128) void wmma_gemm(
    const __bf16* __restrict__ A,
    const __bf16* __restrict__ Wt,
    const float*  __restrict__ inpAdd,
    const float*  __restrict__ bias,
    float* __restrict__ outRaw,
    float* __restrict__ outRelu,
    int M)
{
  constexpr int KSTEPS = KPAD / 32;
  const int lane = threadIdx.x & 31;
  const int wave = threadIdx.x >> 5;
  const int mt0  = blockIdx.x * 8 + wave * 2;   // first of two 16-row M tiles
  const int nt0  = blockIdx.y * 4;              // four 16-col N tiles (always valid)

  v8f acc[2][4];
  #pragma unroll
  for (int mi = 0; mi < 2; ++mi)
    #pragma unroll
    for (int i = 0; i < 4; ++i)
      #pragma unroll
      for (int e = 0; e < 8; ++e) acc[mi][i][e] = 0.0f;

  // ISA A-layout (16x32 bf16): lane<16 -> K[0..7],K[16..23]; lane>=16 -> +8
  const int arow = mt0 * 16 + (lane & 15);
  const int aoff = (lane >> 4) * 8;
  const __bf16* aBase0 = A + (size_t)arow * KPAD + aoff;
  const __bf16* aBase1 = aBase0 + (size_t)16 * KPAD;
  // ISA B-layout (32x16 bf16): lane&15 = column, lane<16 -> K[0..7],K[16..23]; lane>=16 -> +8
  const __bf16* bBase = Wt + (size_t)(nt0 * 16 + (lane & 15)) * KPAD + (lane >> 4) * 8;

  #pragma unroll
  for (int kk = 0; kk < KSTEPS; ++kk) {
    const int ko = kk * 32;
    __builtin_prefetch(aBase0 + ko + 64, 0, 1);   // speculative: safe past end
    __builtin_prefetch(aBase1 + ko + 64, 0, 1);

    v8bf a0lo = *(const v8bf*)(aBase0 + ko);
    v8bf a0hi = *(const v8bf*)(aBase0 + ko + 16);
    v8bf a1lo = *(const v8bf*)(aBase1 + ko);
    v8bf a1hi = *(const v8bf*)(aBase1 + ko + 16);
    v16bf a0 = __builtin_shufflevector(a0lo, a0hi, 0,1,2,3,4,5,6,7,8,9,10,11,12,13,14,15);
    v16bf a1 = __builtin_shufflevector(a1lo, a1hi, 0,1,2,3,4,5,6,7,8,9,10,11,12,13,14,15);

    #pragma unroll
    for (int i = 0; i < 4; ++i) {
      const __bf16* bp = bBase + (size_t)i * 16 * KPAD + ko;
      v8bf blo = *(const v8bf*)bp;
      v8bf bhi = *(const v8bf*)(bp + 16);
      v16bf bf = __builtin_shufflevector(blo, bhi, 0,1,2,3,4,5,6,7,8,9,10,11,12,13,14,15);
      acc[0][i] = __builtin_amdgcn_wmma_f32_16x16x32_bf16(
          false, a0, false, bf, (short)0, acc[0][i], false, false);
      acc[1][i] = __builtin_amdgcn_wmma_f32_16x16x32_bf16(
          false, a1, false, bf, (short)0, acc[1][i], false, false);
    }
  }

  // C/D layout: VGPR r of lane l -> D[mtile*16 + 8*(l>>4) + r][nt*16 + (l&15)]
  const int rowOff  = (lane >> 4) << 3;
  const int colBase = lane & 15;
  #pragma unroll
  for (int mi = 0; mi < 2; ++mi) {
    const int rowBase = (mt0 + mi) * 16 + rowOff;
    #pragma unroll
    for (int i = 0; i < 4; ++i) {
      const int col = (nt0 + i) * 16 + colBase;
      if (col >= HIDDEN) continue;                  // pad columns
      const float bv = (MODE == 2) ? bias[col] : 0.0f;
      #pragma unroll
      for (int r = 0; r < 8; ++r) {
        const int row = rowBase + r;
        if (row >= M) continue;                     // pad rows (last block)
        const size_t o = (size_t)row * HIDDEN + col;
        const float v = acc[mi][i][r];
        if (MODE == 0)      { outRaw[o] = v; outRelu[o] = fmaxf(v, 0.0f); }
        else if (MODE == 1) { outRelu[o] = fmaxf(inpAdd[o] + v, 0.0f); }
        else                { outRelu[o] = fmaxf(v + bv, 0.0f); }
      }
    }
  }
}

// ========== weight convert + transpose:  Wt[n][k] = W[k][n]  (bf16) ==========
__global__ void cvt_weight(const float* __restrict__ W, __bf16* __restrict__ Wt,
                           int K, int N, int kpad)
{
  const int idx = blockIdx.x * blockDim.x + threadIdx.x;
  if (idx >= NPAD * kpad) return;
  const int n = idx / kpad, k = idx % kpad;
  const float v = (n < N && k < K) ? W[(size_t)k * N + n] : 0.0f;
  Wt[idx] = (__bf16)v;
}

// ========== f_bonds (f32, 147 cols) -> bf16 padded [N_BONDS x 160] ==========
__global__ void cvt_fbonds(const float* __restrict__ fb, __bf16* __restrict__ out)
{
  const int idx = blockIdx.x * blockDim.x + threadIdx.x;
  if (idx >= N_BONDS * (KP_I / 4)) return;
  const int b = idx / (KP_I / 4);
  const int k = (idx % (KP_I / 4)) * 4;
  v4bf o;
  #pragma unroll
  for (int e = 0; e < 4; ++e) {
    const int kk = k + e;
    o[e] = (__bf16)((kk < BOND_FDIM) ? fb[(size_t)b * BOND_FDIM + kk] : 0.0f);
  }
  *(v4bf*)(out + (size_t)b * KP_I + k) = o;
}

// ========== a_msg[a] = sum_j message[a2b[a][j]]  (f32, vectorized 16B) ==========
__global__ void gather_atoms(const float* __restrict__ msg, const int* __restrict__ a2b,
                             float* __restrict__ amsg)
{
  const int idx = blockIdx.x * blockDim.x + threadIdx.x;
  if (idx >= N_ATOMS * (HIDDEN / 4)) return;
  const int a = idx / (HIDDEN / 4);
  const int c = (idx % (HIDDEN / 4)) * 4;
  v4f s; s[0] = 0.f; s[1] = 0.f; s[2] = 0.f; s[3] = 0.f;
  #pragma unroll
  for (int j = 0; j < MAX_NB; ++j) {
    const int b = a2b[a * MAX_NB + j];
    s += *(const v4f*)(msg + (size_t)b * HIDDEN + c);
  }
  *(v4f*)(amsg + (size_t)a * HIDDEN + c) = s;
}

// ========== msg_bf16[b] = bf16( a_msg[b2a[b]] - message[b2revb[b]] ), padded ==========
__global__ void build_msg(const float* __restrict__ amsg, const float* __restrict__ msg,
                          const int* __restrict__ b2a, const int* __restrict__ b2revb,
                          __bf16* __restrict__ out)
{
  const int idx = blockIdx.x * blockDim.x + threadIdx.x;
  if (idx >= N_BONDS * (KP_H / 4)) return;
  const int b = idx / (KP_H / 4);
  const int k = (idx % (KP_H / 4)) * 4;
  v4bf o;
  if (k < HIDDEN) {
    const int a  = b2a[b];
    const int rb = b2revb[b];
    const v4f va = *(const v4f*)(amsg + (size_t)a  * HIDDEN + k);
    const v4f vr = *(const v4f*)(msg  + (size_t)rb * HIDDEN + k);
    const v4f d = va - vr;
    #pragma unroll
    for (int e = 0; e < 4; ++e) o[e] = (__bf16)d[e];
  } else {
    #pragma unroll
    for (int e = 0; e < 4; ++e) o[e] = (__bf16)0.0f;
  }
  *(v4bf*)(out + (size_t)b * KP_H + k) = o;
}

// ========== a_input = concat(f_atoms, a_msg) -> bf16 padded [.. x 448] ==========
__global__ void build_ainput(const float* __restrict__ fa, const float* __restrict__ amsg,
                             __bf16* __restrict__ out)
{
  const int idx = blockIdx.x * blockDim.x + threadIdx.x;
  if (idx >= N_ATOMS * (KP_O / 4)) return;
  const int a = idx / (KP_O / 4);
  const int k = (idx % (KP_O / 4)) * 4;
  v4bf o;
  #pragma unroll
  for (int e = 0; e < 4; ++e) {
    const int kk = k + e;
    float v;
    if (kk < ATOM_FDIM)                  v = fa[(size_t)a * ATOM_FDIM + kk];
    else if (kk < ATOM_FDIM + HIDDEN)    v = amsg[(size_t)a * HIDDEN + (kk - ATOM_FDIM)];
    else                                 v = 0.0f;
    o[e] = (__bf16)v;
  }
  *(v4bf*)(out + (size_t)a * KP_O + k) = o;
}

// ========== per-mol atom counts (atom2mol is sorted -> binary search) ==========
__global__ void counts_kernel(const int* __restrict__ a2m, float* __restrict__ counts)
{
  const int m = blockIdx.x * blockDim.x + threadIdx.x;
  if (m >= N_MOLS) return;
  int lo = 0, hi = N_ATOMS;
  while (lo < hi) { int mid = (lo + hi) >> 1; if (a2m[mid] < m) lo = mid + 1; else hi = mid; }
  const int first = lo;
  lo = first; hi = N_ATOMS;
  while (lo < hi) { int mid = (lo + hi) >> 1; if (a2m[mid] < m + 1) lo = mid + 1; else hi = mid; }
  counts[m] = (float)(lo - first);
}

__global__ void zero_f32(float* __restrict__ p, int n)
{
  const int idx = blockIdx.x * blockDim.x + threadIdx.x;
  if (idx < n) p[idx] = 0.0f;
}

// ========== segment-sum pooling: sorted runs accumulated in registers ==========
#define POOL_ATOMS 512
__global__ __launch_bounds__(320) void pool_kernel(const float* __restrict__ hid,
    const int* __restrict__ a2m, float* __restrict__ sums)
{
  const int c = threadIdx.x;
  if (c >= HIDDEN) return;
  const int a0 = blockIdx.x * POOL_ATOMS;
  int a1 = a0 + POOL_ATOMS; if (a1 > N_ATOMS) a1 = N_ATOMS;
  int cur = a2m[a0];
  float acc = 0.0f;
  for (int a = a0; a < a1; ++a) {
    const int m = a2m[a];
    if (m != cur) { atomicAdd(&sums[(size_t)cur * HIDDEN + c], acc); acc = 0.0f; cur = m; }
    acc += hid[(size_t)a * HIDDEN + c];
  }
  atomicAdd(&sums[(size_t)cur * HIDDEN + c], acc);
}

__global__ void finalize(float* __restrict__ out, const float* __restrict__ counts)
{
  const int idx = blockIdx.x * blockDim.x + threadIdx.x;
  if (idx >= N_MOLS * HIDDEN) return;
  const float c = counts[idx / HIDDEN];
  out[idx] = (c > 0.0f) ? out[idx] / c : 0.0f;
}

// =====================================================================
extern "C" void kernel_launch(void* const* d_in, const int* in_sizes, int n_in,
                              void* d_out, int out_size, void* d_ws, size_t ws_size,
                              hipStream_t stream)
{
  const float* f_atoms  = (const float*)d_in[0];
  const float* f_bonds  = (const float*)d_in[1];
  const float* W_i      = (const float*)d_in[2];
  const float* W_h      = (const float*)d_in[3];
  const float* W_o      = (const float*)d_in[4];
  const float* b_o      = (const float*)d_in[5];
  const int*   a2b      = (const int*)d_in[6];
  const int*   b2a      = (const int*)d_in[7];
  const int*   b2revb   = (const int*)d_in[8];
  const int*   atom2mol = (const int*)d_in[9];
  float* out = (float*)d_out;

  // workspace carve-out (256B aligned)
  char* ws = (char*)d_ws;
  size_t off = 0;
  auto take = [&](size_t bytes) -> char* {
    char* p = ws + off;
    off = (off + bytes + 255) & ~(size_t)255;
    return p;
  };
  float*  inp    = (float*)take((size_t)N_BONDS * HIDDEN * 4);   // pre-relu W_i output
  float*  msg    = (float*)take((size_t)N_BONDS * HIDDEN * 4);   // message (f32)
  float*  amsg   = (float*)take((size_t)N_ATOMS * HIDDEN * 4);   // atom agg / final hiddens
  __bf16* abf    = (__bf16*)take((size_t)N_BONDS * KP_H * 2);    // shared bf16 A buffer
                                                                 // (also fits 200064x448)
  __bf16* wti    = (__bf16*)take((size_t)NPAD * KP_I * 2);
  __bf16* wth    = (__bf16*)take((size_t)NPAD * KP_H * 2);
  __bf16* wto    = (__bf16*)take((size_t)NPAD * KP_O * 2);
  float*  counts = (float*)take((size_t)N_MOLS * 4);

  const int T = 256;
  auto cdiv = [](long a, long b) { return (unsigned)((a + b - 1) / b); };

  // weights -> transposed bf16 (tiny, cached in L2 during GEMMs)
  cvt_weight<<<cdiv((long)NPAD * KP_I, T), T, 0, stream>>>(W_i, wti, BOND_FDIM, HIDDEN, KP_I);
  cvt_weight<<<cdiv((long)NPAD * KP_H, T), T, 0, stream>>>(W_h, wth, HIDDEN, HIDDEN, KP_H);
  cvt_weight<<<cdiv((long)NPAD * KP_O, T), T, 0, stream>>>(W_o, wto, ATOM_FDIM + HIDDEN, HIDDEN, KP_O);

  // GEMM1: inp = f_bonds @ W_i ; msg = relu(inp)
  cvt_fbonds<<<cdiv((long)N_BONDS * (KP_I / 4), T), T, 0, stream>>>(f_bonds, abf);
  dim3 gB(cdiv(N_BONDS, 128), 5);      // 128 rows per block (4 waves x 2 Mtiles)
  wmma_gemm<KP_I, 0><<<gB, 128, 0, stream>>>(abf, wti, nullptr, nullptr, inp, msg, N_BONDS);

  // depth-1 = 2 message-passing iterations
  for (int it = 0; it < 2; ++it) {
    gather_atoms<<<cdiv((long)N_ATOMS * (HIDDEN / 4), T), T, 0, stream>>>(msg, a2b, amsg);
    build_msg<<<cdiv((long)N_BONDS * (KP_H / 4), T), T, 0, stream>>>(amsg, msg, b2a, b2revb, abf);
    wmma_gemm<KP_H, 1><<<gB, 128, 0, stream>>>(abf, wth, inp, nullptr, nullptr, msg, N_BONDS);
  }

  // readout: a_input = [f_atoms, a_msg] ; hid = relu(a_input @ W_o + b_o)
  gather_atoms<<<cdiv((long)N_ATOMS * (HIDDEN / 4), T), T, 0, stream>>>(msg, a2b, amsg);
  build_ainput<<<cdiv((long)N_ATOMS * (KP_O / 4), T), T, 0, stream>>>(f_atoms, amsg, abf);
  dim3 gA(cdiv(N_ATOMS, 128), 5);
  wmma_gemm<KP_O, 2><<<gA, 128, 0, stream>>>(abf, wto, nullptr, b_o, nullptr, amsg, N_ATOMS);

  // per-molecule mean pooling
  zero_f32<<<cdiv((long)N_MOLS * HIDDEN, T), T, 0, stream>>>(out, N_MOLS * HIDDEN);
  counts_kernel<<<cdiv((long)N_MOLS, T), T, 0, stream>>>(atom2mol, counts);
  pool_kernel<<<cdiv((long)N_ATOMS, POOL_ATOMS), 320, 0, stream>>>(amsg, atom2mol, out);
  finalize<<<cdiv((long)N_MOLS * HIDDEN, T), T, 0, stream>>>(out, counts);
}
